// MultiLayerLSTM_59966333386917
// MI455X (gfx1250) — compile-verified, matrix operand order checked by reading, not run on hardware
//
#include <hip/hip_runtime.h>
#include <hip/hip_bf16.h>

// Problem constants (match reference)
#define B_   64
#define S_   512
#define E_   512
#define H_   1024
#define G4H  (4 * H_)
#define NBLK 64           // one workgroup per hidden col-tile; 4 waves = 4 batch row-tiles

typedef __attribute__((ext_vector_type(16))) __bf16       v16bf;
typedef __attribute__((ext_vector_type(8)))  float        v8f;
typedef __attribute__((ext_vector_type(8)))  unsigned int v8u;
typedef __attribute__((ext_vector_type(4)))  unsigned int v4u;
typedef __attribute__((ext_vector_type(8)))  int          v8i_;
typedef __attribute__((ext_vector_type(4)))  int          v4i_;

// ---------------------------------------------------------------------------
// scalar helpers
// ---------------------------------------------------------------------------
__device__ __forceinline__ unsigned short f2bf_rne(float f) {
    unsigned int u = __float_as_uint(f);
    u += 0x7FFFu + ((u >> 16) & 1u);
    return (unsigned short)(u >> 16);
}
// two f32 -> packed bf16 (truncate) in ONE v_perm_b32
__device__ __forceinline__ unsigned int pack2_trunc(float lo, float hi) {
    return __builtin_amdgcn_perm(__float_as_uint(hi), __float_as_uint(lo), 0x07060302u);
}
__device__ __forceinline__ unsigned int pack2_rne(float lo, float hi) {
    return (unsigned int)f2bf_rne(lo) | ((unsigned int)f2bf_rne(hi) << 16);
}
__device__ __forceinline__ float bf2f(unsigned short v) {
    return __uint_as_float((unsigned int)v << 16);
}
__device__ __forceinline__ float sigmoidf_(float x) {
    return 1.0f / (1.0f + __expf(-x));
}

// ---------------------------------------------------------------------------
// WMMA operand loaders (16-bit striping per ISA 7.12.2: lanes 0-15 hold
// K = {k0..k0+7, k0+16..k0+23}; lanes 16-31 hold K = {k0+8..+15, k0+24..+31})
// ---------------------------------------------------------------------------
__device__ __forceinline__ v16bf load_tile32B(const unsigned short* __restrict__ p) {
    const uint4* q = reinterpret_cast<const uint4*>(p);
    uint4 q0 = q[0], q1 = q[1];
    v8u v;
    v[0] = q0.x; v[1] = q0.y; v[2] = q0.z; v[3] = q0.w;
    v[4] = q1.x; v[5] = q1.y; v[6] = q1.z; v[7] = q1.w;
    return __builtin_bit_cast(v16bf, v);
}
__device__ __forceinline__ v16bf load_bf_tile(const unsigned short* __restrict__ base,
                                              int ld, int row, int k0, int half) {
    const uint4* p0 = reinterpret_cast<const uint4*>(base + (size_t)row * ld + k0 + half * 8);
    const uint4* p1 = reinterpret_cast<const uint4*>(base + (size_t)row * ld + k0 + 16 + half * 8);
    uint4 q0 = *p0, q1 = *p1;
    v8u v;
    v[0] = q0.x; v[1] = q0.y; v[2] = q0.z; v[3] = q0.w;
    v[4] = q1.x; v[5] = q1.y; v[6] = q1.z; v[7] = q1.w;
    return __builtin_bit_cast(v16bf, v);
}
__device__ __forceinline__ v16bf load_emb_tile(const float* __restrict__ er, int k0, int half) {
    const float4* p0 = reinterpret_cast<const float4*>(er + k0 + half * 8);
    const float4* p1 = reinterpret_cast<const float4*>(er + k0 + half * 8 + 4);
    const float4* p2 = reinterpret_cast<const float4*>(er + k0 + 16 + half * 8);
    const float4* p3 = reinterpret_cast<const float4*>(er + k0 + 16 + half * 8 + 4);
    float4 a = *p0, b = *p1, c = *p2, d = *p3;
    v8u v;
    v[0] = pack2_trunc(a.x, a.y); v[1] = pack2_trunc(a.z, a.w);
    v[2] = pack2_trunc(b.x, b.y); v[3] = pack2_trunc(b.z, b.w);
    v[4] = pack2_trunc(c.x, c.y); v[5] = pack2_trunc(c.z, c.w);
    v[6] = pack2_trunc(d.x, d.y); v[7] = pack2_trunc(d.z, d.w);
    return __builtin_bit_cast(v16bf, v);
}

#define WMMA_BF16(A, Bm, C) \
    __builtin_amdgcn_wmma_f32_16x16x32_bf16(false, (A), false, (Bm), (short)0, (C), false, false)

// ---------------------------------------------------------------------------
// TDM: asynchronously DMA one 4 KB packed-weight chunk (global -> LDS).
// D# per ISA 8.3/8.4: 1-D tile, data_size=8B, tile_dim0=512 elements.
// Arity hedged: ROCm7.2/clang-22 = 5 args, clang-23 toolchain = 6 args.
// ---------------------------------------------------------------------------
__device__ __forceinline__ void tdm_copy_4k(const unsigned short* __restrict__ gsrc,
                                            unsigned int lds_off) {
    unsigned long long ga = (unsigned long long)(uintptr_t)gsrc;
    v4u g0;
    g0[0] = 1u;                                              // count=1, user D#
    g0[1] = lds_off;                                         // lds_addr
    g0[2] = (unsigned int)(ga & 0xFFFFFFFFu);                // global_addr[31:0]
    g0[3] = (unsigned int)((ga >> 32) & 0x01FFFFFFu) | (2u << 30); // addr[56:32]|type=2
    v8i_ g1 = {0, 0, 0, 0, 0, 0, 0, 0};
    g1[0] = (int)(3u << 16);                                 // data_size = 8 bytes
    g1[1] = (int)(512u << 16);                               // tensor_dim0 = 512
    g1[2] = (int)(1u << 16);                                 // tensor_dim1 = 1
    g1[3] = (int)(512u << 16);                               // tile_dim0 = 512 (4 KB)
    g1[4] = 1;                                               // tile_dim1 = 1
    g1[5] = 512;                                             // tensor_dim0_stride
    v4i_ gz = {0, 0, 0, 0};
#if defined(__clang_major__) && (__clang_major__ >= 23)
    v8i_ gz8 = {0, 0, 0, 0, 0, 0, 0, 0};
    __builtin_amdgcn_tensor_load_to_lds(g0, g1, gz, gz, gz8, 0);
#else
    __builtin_amdgcn_tensor_load_to_lds(g0, g1, gz, gz, 0);
#endif
}

// ---------------------------------------------------------------------------
// One GEMM phase with LDS-staged weights.
// Wave 0 TDM-streams 4 KB weight chunks into a 2x4KB LDS double buffer; all
// 4 waves consume the same chunk (one per batch row-tile).  A tiles are
// software-pipelined one chunk ahead from L2-resident global memory.
// ---------------------------------------------------------------------------
template <bool EMBED>
__device__ __forceinline__ void gemm_phase_lds(v8f& accf, v8f& acci, v8f& accg, v8f& acco,
                                               const float* __restrict__ er,
                                               const unsigned short* __restrict__ a1, int arow,
                                               const unsigned short* __restrict__ wstream,
                                               int NC,
                                               unsigned short* __restrict__ smem,
                                               unsigned int smem_off,
                                               int half, int wave, int lane) {
    if (wave == 0) tdm_copy_4k(wstream, smem_off);        // chunk 0 -> buf 0

    v16bf A;
    if constexpr (EMBED) A = load_emb_tile(er, 0, half);
    else                 A = load_bf_tile(a1, H_, arow, 0, half);

    for (int kc = 0; kc < NC; ++kc) {
        if (wave == 0) {
            if (kc + 1 < NC) {
                tdm_copy_4k(wstream + (size_t)(kc + 1) * 2048,
                            smem_off + (unsigned)((kc + 1) & 1) * 4096u);
                __builtin_amdgcn_s_wait_tensorcnt((short)1);   // chunk kc landed
            } else {
                __builtin_amdgcn_s_wait_tensorcnt((short)0);   // drain
            }
        }
        __syncthreads();                                   // buf (kc&1) ready for all waves

        v16bf An;
        const bool more = (kc + 1) < NC;
        if (more) {                                        // A pipelined one chunk ahead
            if constexpr (EMBED) An = load_emb_tile(er, (kc + 1) * 32, half);
            else                 An = load_bf_tile(a1, H_, arow, (kc + 1) * 32, half);
        }

        const unsigned short* bb = smem + (size_t)(kc & 1) * 2048 + lane * 16;
        v16bf Bf = load_tile32B(bb + 0 * 512);
        v16bf Bi = load_tile32B(bb + 1 * 512);
        v16bf Bg = load_tile32B(bb + 2 * 512);
        v16bf Bo = load_tile32B(bb + 3 * 512);
        accf = WMMA_BF16(A, Bf, accf);
        acci = WMMA_BF16(A, Bi, acci);
        accg = WMMA_BF16(A, Bg, accg);
        acco = WMMA_BF16(A, Bo, acco);
        if (more) A = An;

        __syncthreads();                                   // all waves done with buf (kc&1)
    }
}

// lane-local gate math + cell update.
// C/D layout: lane holds column N = lane&15; VGPR r holds row M = r + 8*(lane>>4)
__device__ __forceinline__ void cell_update(const v8f& accf, const v8f& acci,
                                            const v8f& accg, const v8f& acco,
                                            float bfv, float biv, float bgv, float bov,
                                            float* __restrict__ c,
                                            unsigned short* __restrict__ hout,
                                            int rb, int col, int half) {
    #pragma unroll
    for (int r = 0; r < 8; ++r) {
        const int row = rb * 16 + r + half * 8;
        const size_t idx = (size_t)row * H_ + col;
        float fg = sigmoidf_(accf[r] + bfv);
        float ig = sigmoidf_(acci[r] + biv);
        float gg = tanhf(accg[r] + bgv);
        float og = sigmoidf_(acco[r] + bov);
        float cn = fg * c[idx] + ig * gg;
        c[idx] = cn;
        hout[idx] = f2bf_rne(og * tanhf(cn));
    }
}

// device-wide generation barrier (arrive counter + gen flag, s_sleep backoff)
__device__ __forceinline__ void grid_barrier(unsigned int* cnt, unsigned int* gen,
                                             unsigned int& myGen) {
    __threadfence();
    const unsigned int target = ++myGen;
    __syncthreads();
    if (threadIdx.x == 0) {
        unsigned int old = __hip_atomic_fetch_add(cnt, 1u, __ATOMIC_ACQ_REL,
                                                  __HIP_MEMORY_SCOPE_AGENT);
        if (old == NBLK - 1) {
            __hip_atomic_store(cnt, 0u, __ATOMIC_RELAXED, __HIP_MEMORY_SCOPE_AGENT);
            __hip_atomic_fetch_add(gen, 1u, __ATOMIC_RELEASE, __HIP_MEMORY_SCOPE_AGENT);
        }
    }
    while (__hip_atomic_load(gen, __ATOMIC_ACQUIRE, __HIP_MEMORY_SCOPE_AGENT) < target)
        __builtin_amdgcn_s_sleep(1);
}

// ---------------------------------------------------------------------------
// Persistent kernel: 64 workgroups (one per hidden col-tile) x 4 waves (one
// per batch row-tile).  TDM streams each weight chunk into LDS ONCE per
// workgroup per step (4x less L2 traffic than per-wave streaming); loops over
// all 512 timesteps with 2 grid barriers per step.
// ---------------------------------------------------------------------------
__global__ __launch_bounds__(128)
void lstm_persistent(const int* __restrict__ x, const float* __restrict__ emb,
                     const unsigned short* __restrict__ Wx0p,
                     const unsigned short* __restrict__ Wh0p,
                     const unsigned short* __restrict__ Wx1p,
                     const unsigned short* __restrict__ Wh1p,
                     const float* __restrict__ b0, const float* __restrict__ b1,
                     unsigned short* __restrict__ h0p0, unsigned short* __restrict__ h0p1,
                     unsigned short* __restrict__ h1p0, unsigned short* __restrict__ h1p1,
                     float* __restrict__ c0, float* __restrict__ c1,
                     unsigned int* __restrict__ barCnt, unsigned int* __restrict__ barGen) {
    __shared__ unsigned short smem[2 * 2048];      // 2 x 4 KB weight-chunk double buffer

    const int cg   = blockIdx.x;                   // hidden col tile 0..63
    const int wave = threadIdx.x >> 5;             // batch row tile 0..3
    const int lane = threadIdx.x & 31;
    const int half = lane >> 4;
    const int l16  = lane & 15;
    const int col  = cg * 16 + l16;
    const int brow = wave * 16 + l16;              // this lane's batch row (A / token)

    // numeric LDS byte offset of the staging buffer (uniform)
    const unsigned int smoff = (unsigned int)(uintptr_t)(void*)&smem[0];

    // per-workgroup packed-weight stream bases (contiguous per cg)
    const unsigned short* wx0 = Wx0p + (size_t)cg * (E_ / 32) * 2048;
    const unsigned short* wh0 = Wh0p + (size_t)cg * (H_ / 32) * 2048;
    const unsigned short* wx1 = Wx1p + (size_t)cg * (H_ / 32) * 2048;
    const unsigned short* wh1 = Wh1p + (size_t)cg * (H_ / 32) * 2048;

    // per-lane fused biases (fixed column)
    const float b0f = b0[0 * H_ + col], b0i = b0[1 * H_ + col];
    const float b0g = b0[2 * H_ + col], b0o = b0[3 * H_ + col];
    const float b1f = b1[0 * H_ + col], b1i = b1[1 * H_ + col];
    const float b1g = b1[2 * H_ + col], b1o = b1[3 * H_ + col];

    unsigned int myGen = 0;

    for (int t = 0; t < S_; ++t) {
        unsigned short* h0cur = (t & 1) ? h0p1 : h0p0;
        unsigned short* h0nxt = (t & 1) ? h0p0 : h0p1;
        unsigned short* h1cur = (t & 1) ? h1p1 : h1p0;
        unsigned short* h1nxt = (t & 1) ? h1p0 : h1p1;

        // ---- layer 0: z = emb[x[:,t]] @ Wx0^T + h0 @ Wh0^T + b ----------
        {
            v8f accf = {}, acci = {}, accg = {}, acco = {};
            const int tok = x[brow * S_ + t];
            const float* er = emb + (size_t)tok * E_;
            gemm_phase_lds<true >(accf, acci, accg, acco, er, nullptr, 0,
                                  wx0, E_ / 32, smem, smoff, half, wave, lane);
            gemm_phase_lds<false>(accf, acci, accg, acco, nullptr, h0cur, brow,
                                  wh0, H_ / 32, smem, smoff, half, wave, lane);
            cell_update(accf, acci, accg, acco, b0f, b0i, b0g, b0o,
                        c0, h0nxt, wave, col, half);
        }
        grid_barrier(barCnt, barGen, myGen);       // h0nxt visible to all col-tiles

        // ---- layer 1: z = h0' @ Wx1^T + h1 @ Wh1^T + b ------------------
        {
            v8f accf = {}, acci = {}, accg = {}, acco = {};
            gemm_phase_lds<false>(accf, acci, accg, acco, nullptr, h0nxt, brow,
                                  wx1, H_ / 32, smem, smoff, half, wave, lane);
            gemm_phase_lds<false>(accf, acci, accg, acco, nullptr, h1cur, brow,
                                  wh1, H_ / 32, smem, smoff, half, wave, lane);
            cell_update(accf, acci, accg, acco, b1f, b1i, b1g, b1o,
                        c1, h1nxt, wave, col, half);
        }
        grid_barrier(barCnt, barGen, myGen);       // h1nxt visible before next step
    }
}

// ---------------------------------------------------------------------------
// prep / epilogue kernels
// ---------------------------------------------------------------------------
// Repack f32 [4H][K] gate-stacked weights into WMMA-tile-major bf16:
// offset = ((cg*(K/32) + kc)*4 + gate)*512 + lane*16  (halfs); 4 KB per chunk
__global__ void k_pack_w(const float* __restrict__ W, unsigned short* __restrict__ Wp, int K) {
    const int nkc = K / 32;
    const int total = (H_ / 16) * nkc * 128;
    int i = blockIdx.x * blockDim.x + threadIdx.x;
    if (i >= total) return;
    const int lane = i & 31;
    const int gate = (i >> 5) & 3;
    const int rest = i >> 7;
    const int kc = rest % nkc;
    const int cg = rest / nkc;
    const int half = lane >> 4, l16 = lane & 15;
    const int row = gate * H_ + cg * 16 + l16;
    const float* src = W + (size_t)row * K + kc * 32;
    unsigned int v[8];
    #pragma unroll
    for (int j = 0; j < 4; ++j) {
        v[j]     = pack2_rne(src[half * 8 + 2 * j],      src[half * 8 + 2 * j + 1]);
        v[4 + j] = pack2_rne(src[16 + half * 8 + 2 * j], src[16 + half * 8 + 2 * j + 1]);
    }
    const size_t slot = (((size_t)cg * nkc + kc) * 4 + gate) * 32 + lane;
    uint4* d = reinterpret_cast<uint4*>(Wp + slot * 16);
    d[0] = make_uint4(v[0], v[1], v[2], v[3]);
    d[1] = make_uint4(v[4], v[5], v[6], v[7]);
}
__global__ void k_bias_combine(const float* __restrict__ a, const float* __restrict__ b,
                               float* __restrict__ d, int n) {
    int i = blockIdx.x * blockDim.x + threadIdx.x;
    if (i < n) d[i] = a[i] + b[i];
}
__global__ void k_zero_u16(unsigned short* __restrict__ d, int n) {
    int i = blockIdx.x * blockDim.x + threadIdx.x;
    if (i < n) d[i] = 0;
}
__global__ void k_zero_f32(float* __restrict__ d, int n) {
    int i = blockIdx.x * blockDim.x + threadIdx.x;
    if (i < n) d[i] = 0.0f;
}
__global__ void k_zero_u32(unsigned int* __restrict__ d, int n) {
    int i = blockIdx.x * blockDim.x + threadIdx.x;
    if (i < n) d[i] = 0u;
}

// out[b] = sigmoid(h1[b,:] . fc_w + fc_b), one block per batch row
__global__ __launch_bounds__(256)
void fc_kernel(const unsigned short* __restrict__ h1, const float* __restrict__ w,
               const float* __restrict__ bptr, float* __restrict__ out) {
    __shared__ float red[8];
    const int b = blockIdx.x;
    float s = 0.0f;
    for (int j = threadIdx.x; j < H_; j += 256)
        s += bf2f(h1[(size_t)b * H_ + j]) * w[j];
    #pragma unroll
    for (int o = 16; o > 0; o >>= 1) s += __shfl_down(s, o, 32);
    if ((threadIdx.x & 31) == 0) red[threadIdx.x >> 5] = s;
    __syncthreads();
    if (threadIdx.x == 0) {
        float t = 0.0f;
        #pragma unroll
        for (int i = 0; i < 8; ++i) t += red[i];
        out[b] = 1.0f / (1.0f + __expf(-(t + bptr[0])));
    }
}

// ---------------------------------------------------------------------------
// host launcher
// ---------------------------------------------------------------------------
static inline size_t align256(size_t v) { return (v + 255) & ~(size_t)255; }

extern "C" void kernel_launch(void* const* d_in, const int* in_sizes, int n_in,
                              void* d_out, int out_size, void* d_ws, size_t ws_size,
                              hipStream_t stream) {
    (void)in_sizes; (void)n_in; (void)out_size; (void)ws_size;

    const int*   x   = (const int*)  d_in[0];
    const float* emb = (const float*)d_in[1];
    const float* Wx0 = (const float*)d_in[2];
    const float* bx0 = (const float*)d_in[3];
    const float* Wh0 = (const float*)d_in[4];
    const float* bh0 = (const float*)d_in[5];
    const float* Wx1 = (const float*)d_in[6];
    const float* bx1 = (const float*)d_in[7];
    const float* Wh1 = (const float*)d_in[8];
    const float* bh1 = (const float*)d_in[9];
    const float* fcw = (const float*)d_in[10];
    const float* fcb = (const float*)d_in[11];
    float*       out = (float*)d_out;

    char* ws = (char*)d_ws;
    size_t off = 0;
    auto carve = [&](size_t bytes) { char* p = ws + off; off = align256(off + bytes); return p; };

    unsigned short* Wx0p = (unsigned short*)carve((size_t)G4H * E_ * 2);
    unsigned short* Wh0p = (unsigned short*)carve((size_t)G4H * H_ * 2);
    unsigned short* Wx1p = (unsigned short*)carve((size_t)G4H * H_ * 2);
    unsigned short* Wh1p = (unsigned short*)carve((size_t)G4H * H_ * 2);
    float*          b0   = (float*)carve((size_t)G4H * 4);
    float*          b1   = (float*)carve((size_t)G4H * 4);
    unsigned short* h0p0 = (unsigned short*)carve((size_t)B_ * H_ * 2);
    unsigned short* h0p1 = (unsigned short*)carve((size_t)B_ * H_ * 2);
    unsigned short* h1p0 = (unsigned short*)carve((size_t)B_ * H_ * 2);
    unsigned short* h1p1 = (unsigned short*)carve((size_t)B_ * H_ * 2);
    float*          c0   = (float*)carve((size_t)B_ * H_ * 4);
    float*          c1   = (float*)carve((size_t)B_ * H_ * 4);
    unsigned int*   bar  = (unsigned int*)carve(2 * sizeof(unsigned int));

    const int TPB = 256;
    auto blks = [&](int n) { return (n + TPB - 1) / TPB; };

    // per-launch prep: repack weights (bf16, tile-major), fuse biases, init state
    const int npack_E = (H_ / 16) * (E_ / 32) * 128;
    const int npack_H = (H_ / 16) * (H_ / 32) * 128;
    k_pack_w<<<blks(npack_E), TPB, 0, stream>>>(Wx0, Wx0p, E_);
    k_pack_w<<<blks(npack_H), TPB, 0, stream>>>(Wh0, Wh0p, H_);
    k_pack_w<<<blks(npack_H), TPB, 0, stream>>>(Wx1, Wx1p, H_);
    k_pack_w<<<blks(npack_H), TPB, 0, stream>>>(Wh1, Wh1p, H_);
    k_bias_combine<<<blks(G4H), TPB, 0, stream>>>(bx0, bh0, b0, G4H);
    k_bias_combine<<<blks(G4H), TPB, 0, stream>>>(bx1, bh1, b1, G4H);
    k_zero_u16<<<blks(B_ * H_), TPB, 0, stream>>>(h0p0, B_ * H_);
    k_zero_u16<<<blks(B_ * H_), TPB, 0, stream>>>(h1p0, B_ * H_);
    k_zero_f32<<<blks(B_ * H_), TPB, 0, stream>>>(c0, B_ * H_);
    k_zero_f32<<<blks(B_ * H_), TPB, 0, stream>>>(c1, B_ * H_);
    k_zero_u32<<<1, 32, 0, stream>>>(bar, 2);

    // whole 512-step scan in ONE persistent kernel (2 grid barriers / step)
    lstm_persistent<<<NBLK, 128, 0, stream>>>(x, emb, Wx0p, Wh0p, Wx1p, Wh1p, b0, b1,
                                              h0p0, h0p1, h1p0, h1p1, c0, c1,
                                              bar + 0, bar + 1);

    // final projection (final h1 lives in buffer 0 after 512 steps)
    fc_kernel<<<B_, 256, 0, stream>>>(h1p0, fcw, fcb, out);
}
